// CRF_model_31061203484726
// MI455X (gfx1250) — compile-verified, matrix operand order checked by reading, not run on hardware
//
#include <hip/hip_runtime.h>
#include <cstdint>

#define CRF_B 256
#define CRF_S 2048
#define CRF_T 32

typedef float    v2f  __attribute__((ext_vector_type(2)));
typedef float    v8f  __attribute__((ext_vector_type(8)));
typedef _Float16 v16h __attribute__((ext_vector_type(16)));

#if defined(__has_builtin)
#if __has_builtin(__builtin_amdgcn_wmma_f32_16x16x4_f32)
#define CRF_HAVE_WMMA_F32X4 1
#endif
#endif

// ---------------------------------------------------------------------------
// Viterbi: 1 wave per batch, lane j == tag j. Max-plus semiring -> VALU only.
// Backpointer history (2047 x 32 bytes = 65504 B) lives entirely in LDS
// (CDNA5 WGP has 320KB), backtrace done in-kernel -> no HBM history traffic.
// ---------------------------------------------------------------------------
__global__ __launch_bounds__(32) void crf_viterbi_kernel(
    const float* __restrict__ pred, const int* __restrict__ mask,
    const float* __restrict__ startt, const float* __restrict__ endt,
    const float* __restrict__ trans, float* __restrict__ out)
{
    __shared__ unsigned char bp[(CRF_S - 1) * CRF_T];   // 65504 bytes
    const int b = blockIdx.x;
    const int j = threadIdx.x;                          // tag lane (wave32)

    // Column j of the transition matrix, kept in registers (static indexing).
    float tcol[CRF_T];
#pragma unroll
    for (int i = 0; i < CRF_T; ++i) tcol[i] = trans[i * CRF_T + j];

    float score = startt[j] + pred[((size_t)b * CRF_S) * CRF_T + j];

    for (int s = 1; s < CRF_S; ++s) {
        if (s + 8 < CRF_S)
            __builtin_prefetch(&pred[((size_t)b * CRF_S + s + 8) * CRF_T + j], 0, 0);
        float em = pred[((size_t)b * CRF_S + s) * CRF_T + j];
        int   mk = mask[(size_t)b * CRF_S + s];

        float best = -3.402823466e38f;
        int   bidx = 0;
#pragma unroll
        for (int i = 0; i < CRF_T; ++i) {
            float si   = __shfl(score, i, 32);          // v_readlane broadcast
            float cand = si + tcol[i];
            if (cand > best) { best = cand; bidx = i; } // strict > == first argmax
        }
        score = mk ? (best + em) : score;
        bp[(s - 1) * CRF_T + j] = (unsigned char)(mk ? bidx : j);
    }

    // Cross-lane argmax of score + end (min-index tie break == jnp.argmax).
    float fin  = score + endt[j];
    int   ftag = j;
#pragma unroll
    for (int off = 16; off > 0; off >>= 1) {
        float ov = __shfl_xor(fin,  off, 32);
        int   ot = __shfl_xor(ftag, off, 32);
        if (ov > fin || (ov == fin && ot < ftag)) { fin = ov; ftag = ot; }
    }

    if (j == 0) {
        int tag = ftag;
        for (int s = CRF_S - 1; s >= 1; --s) {
            out[(size_t)b * CRF_S + s] = (float)tag;
            tag = (int)bp[(s - 1) * CRF_T + tag];
        }
        out[(size_t)b * CRF_S + 0] = (float)tag;
    }
}

// ---------------------------------------------------------------------------
// Numerator: gather-sum, 1 wave per batch, lanes stride over s.
// ---------------------------------------------------------------------------
__global__ __launch_bounds__(32) void crf_numerator_kernel(
    const float* __restrict__ pred, const int* __restrict__ mask,
    const int* __restrict__ labels, const float* __restrict__ startt,
    const float* __restrict__ endt, const float* __restrict__ trans,
    float* __restrict__ num)
{
    const int b = blockIdx.x;
    const int l = threadIdx.x;
    float acc = 0.f;
    int seqlen = 0;
    for (int s = l; s < CRF_S; s += 32) {
        int mk = mask[(size_t)b * CRF_S + s];
        seqlen += mk;
        if (s >= 1 && mk) {
            int t0 = labels[(size_t)b * CRF_S + s - 1];
            int t1 = labels[(size_t)b * CRF_S + s];
            acc += trans[t0 * CRF_T + t1] + pred[((size_t)b * CRF_S + s) * CRF_T + t1];
        }
    }
#pragma unroll
    for (int off = 16; off > 0; off >>= 1) {
        acc    += __shfl_xor(acc,    off, 32);
        seqlen += __shfl_xor(seqlen, off, 32);
    }
    if (l == 0) {
        int   t0 = labels[(size_t)b * CRF_S];
        float sc = startt[t0] + pred[((size_t)b * CRF_S) * CRF_T + t0] + acc;
        int   lt = labels[(size_t)b * CRF_S + (seqlen - 1)];
        num[b] = sc + endt[lt];
    }
}

// ---------------------------------------------------------------------------
// Denominator: shifted-exp forward algorithm on the matrix cores.
//   score'[b,j] = m_b + log( sum_i exp(score[b,i]-m_b) * exp(trans[i,j]) ) + em[b,j]
// 16 batches per wave: alpha(16x32) @ expTrans(32x32) via 16 chained
// v_wmma_f32_16x16x4_f32 (2 N-halves x 8 K-chunks), fp32-exact logsumexp.
//
// Fragment layouts (ISA 7.12.2, 32-bit 16x4 A / 4x16 B / 16x16 C):
//   A: lane l, m = l%16, VGPR0 -> K = 4kc + (l<16 ? 0 : 2), VGPR1 -> +1
//   B: lane l, n = 16h + l%16, same K striping as A
//   C: VGPR r, lanes 0-15 -> (m=r, n), lanes 16-31 -> (m=r+8, n)
//
// NOTE: all register-array indices below are compile-time constants; the
// hiHalf lane split is a single v_cndmask select, never a dynamic register
// index (which previously lowered as ~32-way cmp/cndmask chains).
// ---------------------------------------------------------------------------
__global__ __launch_bounds__(32) void crf_denominator_kernel(
    const float* __restrict__ pred, const int* __restrict__ mask,
    const float* __restrict__ startt, const float* __restrict__ endt,
    const float* __restrict__ trans, float* __restrict__ den)
{
    __shared__ float sc[16 * CRF_T];    // log-domain scores, row-major [m][n]
    __shared__ float mbuf[16];          // per-batch max shift

    const int  l      = threadIdx.x;
    const int  lo     = l & 15;
    const bool hiHalf = (l >= 16);
    const int  b0     = blockIdx.x * 16;
    const int  d0     = hiHalf ? 2 : 0;

    // Precompute B-fragments of E = exp(trans): Bf[kc][h], constant over s.
#if defined(CRF_HAVE_WMMA_F32X4)
    v2f Bf[8][2];
#pragma unroll
    for (int kc = 0; kc < 8; ++kc)
#pragma unroll
        for (int h = 0; h < 2; ++h) {
            int n = 16 * h + lo;
            Bf[kc][h].x = __expf(trans[(4 * kc + d0)     * CRF_T + n]);
            Bf[kc][h].y = __expf(trans[(4 * kc + d0 + 1) * CRF_T + n]);
        }
#else
    // Fallback: f16 WMMA (codegen-confirmed builtin), one 16x16x32 per N-half.
    v16h Bh[2];
#pragma unroll
    for (int h = 0; h < 2; ++h)
#pragma unroll
        for (int e = 0; e < 16; ++e) {
            int p = e >> 1, off = e & 1;
            int k = ((p < 4) ? (2 * p) : (8 + 2 * p)) + (hiHalf ? 8 : 0) + off;
            Bh[h][e] = (_Float16)__expf(trans[k * CRF_T + (16 * h + lo)]);
        }
#endif

    // score0[m][n] = start[n] + em[m][0][n]; lane l writes column n = l.
    for (int m = 0; m < 16; ++m)
        sc[m * CRF_T + l] = startt[l] + pred[((size_t)(b0 + m) * CRF_S) * CRF_T + l];
    __syncthreads();

    for (int s = 1; s < CRF_S; ++s) {
        // Each lane pulls its batch row (b = lo) from LDS: 8x ds_load_b128.
        float row[CRF_T];
#pragma unroll
        for (int q4 = 0; q4 < 8; ++q4) {
            float4 v = *(const float4*)&sc[lo * CRF_T + 4 * q4];
            row[4 * q4 + 0] = v.x; row[4 * q4 + 1] = v.y;
            row[4 * q4 + 2] = v.z; row[4 * q4 + 3] = v.w;
        }
        float mb = row[0];
#pragma unroll
        for (int k = 1; k < CRF_T; ++k) mb = fmaxf(mb, row[k]);
        if (!hiHalf) mbuf[lo] = mb;

        // q[16x16] per N-half = exp(row - mb) @ expTrans, on the matrix cores.
        v8f q0 = {}; v8f q1 = {};
#if defined(CRF_HAVE_WMMA_F32X4)
#pragma unroll
        for (int kc = 0; kc < 8; ++kc) {
            // Static indices + one-select lane split (no dynamic reg indexing).
            float r0 = hiHalf ? row[4 * kc + 2] : row[4 * kc + 0];
            float r1 = hiHalf ? row[4 * kc + 3] : row[4 * kc + 1];
            v2f a;
            a.x = __expf(r0 - mb);
            a.y = __expf(r1 - mb);
            q0 = __builtin_amdgcn_wmma_f32_16x16x4_f32(false, a, false, Bf[kc][0],
                                                       (short)0, q0, false, false);
            q1 = __builtin_amdgcn_wmma_f32_16x16x4_f32(false, a, false, Bf[kc][1],
                                                       (short)0, q1, false, false);
        }
#else
        v16h a16;
#pragma unroll
        for (int e = 0; e < 16; ++e) {
            int p = e >> 1, off = e & 1;
            int kb = ((p < 4) ? (2 * p) : (8 + 2 * p)) + off;   // static
            float rv = hiHalf ? row[kb + 8] : row[kb];          // one select
            a16[e] = (_Float16)__expf(rv - mb);
        }
        q0 = __builtin_amdgcn_wmma_f32_16x16x32_f16(false, a16, false, Bh[0],
                                                    (short)0, q0, false, false);
        q1 = __builtin_amdgcn_wmma_f32_16x16x32_f16(false, a16, false, Bh[1],
                                                    (short)0, q1, false, false);
#endif
        __syncthreads();

        // New scores in C layout: lane l owns (m = r + 8*hiHalf, n = lo + 16h).
        float nv[8][2];
#pragma unroll
        for (int r = 0; r < 8; ++r) {
            int m  = r + (hiHalf ? 8 : 0);
            int gb = b0 + m;
            int mk = mask[(size_t)gb * CRF_S + s];
            float mbm = mbuf[m];
#pragma unroll
            for (int h = 0; h < 2; ++h) {
                int   n   = lo + 16 * h;
                float old = sc[m * CRF_T + n];
                float em  = pred[((size_t)gb * CRF_S + s) * CRF_T + n];
                float qv  = h ? q1[r] : q0[r];
                nv[r][h]  = mk ? (mbm + __logf(qv) + em) : old;
            }
        }
        __syncthreads();                        // all old-score reads done
#pragma unroll
        for (int r = 0; r < 8; ++r) {
            int m = r + (hiHalf ? 8 : 0);
#pragma unroll
            for (int h = 0; h < 2; ++h)
                sc[m * CRF_T + lo + 16 * h] = nv[r][h];
        }
        __syncthreads();
    }

    // den[b] = logsumexp_j(score[b,j] + end[j]); lanes 0-15 write.
    float rowf[CRF_T];
#pragma unroll
    for (int q4 = 0; q4 < 8; ++q4) {
        float4 v = *(const float4*)&sc[lo * CRF_T + 4 * q4];
        rowf[4 * q4 + 0] = v.x; rowf[4 * q4 + 1] = v.y;
        rowf[4 * q4 + 2] = v.z; rowf[4 * q4 + 3] = v.w;
    }
    float mb2 = rowf[0] + endt[0];
#pragma unroll
    for (int k = 1; k < CRF_T; ++k) mb2 = fmaxf(mb2, rowf[k] + endt[k]);
    float ssum = 0.f;
#pragma unroll
    for (int k = 0; k < CRF_T; ++k) ssum += __expf(rowf[k] + endt[k] - mb2);
    if (!hiHalf) den[b0 + lo] = mb2 + __logf(ssum);
}

// ---------------------------------------------------------------------------
// loss = -mean(num - den)
// ---------------------------------------------------------------------------
__global__ __launch_bounds__(256) void crf_loss_kernel(
    const float* __restrict__ num, const float* __restrict__ den,
    float* __restrict__ out)
{
    __shared__ float partial[8];
    const int t = threadIdx.x;
    float v = num[t] - den[t];
#pragma unroll
    for (int off = 16; off > 0; off >>= 1) v += __shfl_xor(v, off, 32);
    if ((t & 31) == 0) partial[t >> 5] = v;
    __syncthreads();
    if (t == 0) {
        float sum = 0.f;
#pragma unroll
        for (int i = 0; i < 8; ++i) sum += partial[i];
        out[(size_t)CRF_B * CRF_S] = -sum / (float)CRF_B;
    }
}

// ---------------------------------------------------------------------------
extern "C" void kernel_launch(void* const* d_in, const int* in_sizes, int n_in,
                              void* d_out, int out_size, void* d_ws, size_t ws_size,
                              hipStream_t stream) {
    const float* pred   = (const float*)d_in[0];
    const int*   amask  = (const int*)  d_in[1];
    const int*   labels = (const int*)  d_in[2];
    const float* startt = (const float*)d_in[3];
    const float* endt   = (const float*)d_in[4];
    const float* trans  = (const float*)d_in[5];
    float* out = (float*)d_out;

    float* num = (float*)d_ws;          // [B]
    float* den = num + CRF_B;           // [B]

    crf_viterbi_kernel    <<<CRF_B,      32, 0, stream>>>(pred, amask, startt, endt, trans, out);
    crf_numerator_kernel  <<<CRF_B,      32, 0, stream>>>(pred, amask, labels, startt, endt, trans, num);
    crf_denominator_kernel<<<CRF_B / 16, 32, 0, stream>>>(pred, amask, startt, endt, trans, den);
    crf_loss_kernel       <<<1,         256, 0, stream>>>(num, den, out);
}
